// SpatialDiffAttn_65781719105747
// MI455X (gfx1250) — matrix operand order repaired and checked by reading, not compiled, not run on hardware
//
#include <hip/hip_runtime.h>
#include <hip/hip_bf16.h>

// ---------------------------------------------------------------------------
// SpatialDiffAttn on gfx1250 (MI455X): bf16 WMMA QKV GEMM + flash-style
// differential attention (TDM async K-tile staging into LDS) + fused LayerNorm
// ---------------------------------------------------------------------------

typedef __attribute__((ext_vector_type(16))) __bf16 v16bf;
typedef __attribute__((ext_vector_type(8)))  __bf16 v8bf;
typedef __attribute__((ext_vector_type(8)))  float  v8f;
typedef unsigned int u32x4 __attribute__((ext_vector_type(4)));
typedef int          i32x4 __attribute__((ext_vector_type(4)));
typedef int          i32x8 __attribute__((ext_vector_type(8)));

#define BT_     32      // B*T
#define NTOK_   512     // spatial tokens
#define MROWS_  (BT_*NTOK_)   // 16384
#define HEADS_  8

__constant__ const float kScale      = 0.25f;                 // HD^-0.5
__constant__ const float kLambdaInit = 0.35550906759096927f;  // 0.8-0.6*exp(-0.3)
__constant__ const float kLnEps      = 1e-5f;

#if __has_builtin(__builtin_amdgcn_tensor_load_to_lds) && \
    __has_builtin(__builtin_amdgcn_s_wait_tensorcnt)
#define HAVE_TDM 1
#else
#define HAVE_TDM 0
#endif

// ---- helpers ---------------------------------------------------------------

__device__ __forceinline__ unsigned short f2bf(float f) {
  unsigned int u = __builtin_bit_cast(unsigned int, f);
  u += 0x7fffu + ((u >> 16) & 1u);      // round-to-nearest-even
  return (unsigned short)(u >> 16);
}

__device__ __forceinline__ v8bf ld8(const unsigned short* p) {
  uint4 u = *(const uint4*)p;           // 16B load (global or ds)
  return __builtin_bit_cast(v8bf, u);
}

__device__ __forceinline__ v16bf cat8(v8bf lo, v8bf hi) {
  return __builtin_shufflevector(lo, hi, 0,1,2,3,4,5,6,7,8,9,10,11,12,13,14,15);
}

__device__ __forceinline__ v16bf ld16(const unsigned short* p) {
  return cat8(ld8(p), ld8(p + 8));
}

__device__ __forceinline__ v8bf zero8() {
  return __builtin_bit_cast(v8bf, make_uint4(0u, 0u, 0u, 0u));
}

__device__ __forceinline__ v8f wmma_bf16(v16bf a, v16bf b, v8f c) {
  // D = A(16x32 bf16) x B(32x16 bf16) + C(16x16 f32)
  return __builtin_amdgcn_wmma_f32_16x16x32_bf16(
      /*neg_a=*/false, a, /*neg_b=*/false, b,
      /*c_mod=*/(short)0, c, /*reuse_a=*/false, /*reuse_b=*/false);
}

// reductions across the 16-lane group sharing a C/D-layout row (wave32)
__device__ __forceinline__ float rmax16(float v) {
  v = fmaxf(v, __shfl_xor(v, 1, 32));
  v = fmaxf(v, __shfl_xor(v, 2, 32));
  v = fmaxf(v, __shfl_xor(v, 4, 32));
  v = fmaxf(v, __shfl_xor(v, 8, 32));
  return v;
}
__device__ __forceinline__ float radd16(float v) {
  v += __shfl_xor(v, 1, 32);
  v += __shfl_xor(v, 2, 32);
  v += __shfl_xor(v, 4, 32);
  v += __shfl_xor(v, 8, 32);
  return v;
}
__device__ __forceinline__ float radd32(float v) {
  v = radd16(v);
  v += __shfl_xor(v, 16, 32);
  return v;
}

// ---- TDM: async 32x32 bf16 tile (64B rows, stride 512B) global -> LDS ------
// D# layout per CDNA5 ISA.
__device__ __forceinline__ void tdm_load_tile_2d(unsigned int lds_byte_addr,
                                                 const unsigned short* gsrc) {
#if HAVE_TDM
  unsigned long long ga = (unsigned long long)(uintptr_t)gsrc;
  u32x4 g0;
  g0[0] = 1u;                                             // count=1, user mode
  g0[1] = lds_byte_addr;                                  // lds_addr
  g0[2] = (unsigned int)(ga & 0xFFFFFFFFull);             // global_addr[31:0]
  g0[3] = (unsigned int)((ga >> 32) & 0x1FFFFFFull)       // global_addr[56:32]
          | (2u << 30);                                   // type=2 ("image")
  i32x8 g1;
  g1[0] = 0x00010000;          // wg_mask=0, data_size=1 (2B elements)
  g1[1] = (int)(256u << 16);   // tensor_dim0 = 256 (lo16 at bits [63:48])
  g1[2] = (int)(0x4000u << 16);// tensor_dim0 hi=0; tensor_dim1 = 16384 lo16
  g1[3] = (int)(32u << 16);    // tensor_dim1 hi=0; tile_dim0 = 32
  g1[4] = 32;                  // tile_dim1 = 32, tile_dim2 = 0 (2D)
  g1[5] = 256;                 // tensor_dim0_stride[31:0] = 256 elements
  g1[6] = 0;                   // stride hi, dim1_stride lo
  g1[7] = 0;
  i32x4 g2 = {0, 0, 0, 0};
  i32x4 g3 = {0, 0, 0, 0};
#if defined(__clang_major__) && (__clang_major__ >= 23)
  i32x8 g4 = {0, 0, 0, 0, 0, 0, 0, 0};
  __builtin_amdgcn_tensor_load_to_lds(g0, g1, g2, g3, g4, 0);
#else
  __builtin_amdgcn_tensor_load_to_lds(g0, g1, g2, g3, 0);
#endif
#else
  (void)lds_byte_addr; (void)gsrc;
#endif
}

template <int N>
__device__ __forceinline__ void tdm_wait() {
#if HAVE_TDM
  __builtin_amdgcn_s_wait_tensorcnt(N);   // template arg is an ICE
#endif
}

// ---- stage 0: conversions --------------------------------------------------

__global__ void cvt_x_bf16(const float* __restrict__ x,
                           unsigned short* __restrict__ xb, int n4) {
  int i = blockIdx.x * blockDim.x + threadIdx.x;
  if (i >= n4) return;
  float4 f = ((const float4*)x)[i];
  ushort4 u;
  u.x = f2bf(f.x); u.y = f2bf(f.y); u.z = f2bf(f.z); u.w = f2bf(f.w);
  ((ushort4*)xb)[i] = u;
}

// W [K=256, C=256] -> Wt bf16 [C, K] (contiguous along K for B-operand loads)
__global__ void xpose_w_bf16(const float* __restrict__ wq,
                             const float* __restrict__ wk,
                             const float* __restrict__ wv,
                             unsigned short* __restrict__ wt) {
  int i = blockIdx.x * blockDim.x + threadIdx.x;   // 3*65536 threads
  int mat = i >> 16;
  int rem = i & 65535;
  int c = rem >> 8, k = rem & 255;
  const float* w = (mat == 0) ? wq : (mat == 1) ? wk : wv;
  wt[mat * 65536 + c * 256 + k] = f2bf(w[k * 256 + c]);
}

// ---- stage 1: QKV projection GEMM (bf16 WMMA, f32 accumulate) -------------
// One wave per 16x16 output tile; K = 256 in 8 WMMA steps.
// Q,K  -> bf16 [token, 256]
// V    -> bf16 transposed [bt, h, d(32), m(512)]

__global__ void qkv_gemm(const unsigned short* __restrict__ xb,
                         const unsigned short* __restrict__ wt,
                         unsigned short* __restrict__ qb,
                         unsigned short* __restrict__ kb,
                         unsigned short* __restrict__ vt) {
  const int lane = threadIdx.x;
  const int lm = lane & 15, half = lane >> 4;
  const int mtile = blockIdx.x;    // 0..1023
  const int ctile = blockIdx.y;    // 0..15
  const int mat   = blockIdx.z;    // 0=Q 1=K 2=V
  const int row = mtile * 16 + lm;
  const int col = ctile * 16 + lm;
  const unsigned short* w = wt + mat * 65536;

  v8f acc = {0.f, 0.f, 0.f, 0.f, 0.f, 0.f, 0.f, 0.f};

#pragma unroll
  for (int kt = 0; kt < 8; ++kt) {
    // A (16-bit A layout): lane row=M, slots e<8 -> K=kt*32+8*half+e,
    //                      slots e>=8 -> K=kt*32+16+8*half+(e-8)
    const unsigned short* arow = xb + row * 256 + kt * 32 + 8 * half;
    v16bf A = cat8(ld8(arow), ld8(arow + 16));
    // B (16-bit B layout): lane col=N, slots e<8 -> K=kt*32+8*half+e,
    //                      slots e>=8 -> K=kt*32+16+8*half+(e-8)
    const unsigned short* bcol = w + col * 256 + kt * 32 + 8 * half;
    v16bf B = cat8(ld8(bcol), ld8(bcol + 16));
    if (kt + 1 < 8)
      __builtin_prefetch(xb + row * 256 + (kt + 1) * 32, 0, 1);
    acc = wmma_bf16(A, B, acc);
  }

  if (mat == 2) {
    // transpose-store into Vt[bt, h, d, m]; lane owns fixed (h,d), rows pack
    // into 8 consecutive m -> one 16B store.
    int rbase = mtile * 16 + 8 * half;          // first row this lane holds
    int bt = rbase >> 9;
    int mbase = rbase & 511;
    int h = col >> 5, d = col & 31;
    unsigned short tmp[8];
#pragma unroll
    for (int r = 0; r < 8; ++r) tmp[r] = f2bf(acc[r]);
    uint4 u;
    u.x = (unsigned)tmp[0] | ((unsigned)tmp[1] << 16);
    u.y = (unsigned)tmp[2] | ((unsigned)tmp[3] << 16);
    u.z = (unsigned)tmp[4] | ((unsigned)tmp[5] << 16);
    u.w = (unsigned)tmp[6] | ((unsigned)tmp[7] << 16);
    *(uint4*)(vt + (((size_t)(bt * 8 + h) * 32 + d) << 9) + mbase) = u;
  } else {
    unsigned short* dst = (mat == 0) ? qb : kb;
#pragma unroll
    for (int r = 0; r < 8; ++r) {
      int rout = mtile * 16 + r + 8 * half;     // D layout: M = r + 8*half
      dst[(size_t)rout * 256 + col] = f2bf(acc[r]);
    }
  }
}

// ---- stage 2: differential attention + LayerNorm --------------------------
// One wave per (bt, head, 16-query tile). Flash over 16 iterations x 32 keys.
// K tiles (32 keys x 32 ch = 2KB) are TDM-DMA'd into LDS, double buffered,
// overlapped with the WMMA + softmax work of the previous iteration.

__device__ __forceinline__ void flash_update(v8f sa, v8f sb,
                                             float* mst, float* lst,
                                             v8f& accA, v8f& accB,
                                             unsigned short* Ps,
                                             int lm, int half) {
#pragma unroll
  for (int r = 0; r < 8; ++r) {
    float va = sa[r] * kScale;
    float vb = sb[r] * kScale;
    float t = rmax16(fmaxf(va, vb));
    float mn = fmaxf(mst[r], t);
    float corr = __expf(mst[r] - mn);
    float pa = __expf(va - mn);
    float pb = __expf(vb - mn);
    float rs = radd16(pa + pb);
    lst[r] = lst[r] * corr + rs;
    mst[r] = mn;
    accA[r] *= corr;
    accB[r] *= corr;
    int n = r + 8 * half;                 // D-layout row owned by this lane
    Ps[n * 32 + lm]      = f2bf(pa);      // keys [0,16) of this 32-key block
    Ps[n * 32 + 16 + lm] = f2bf(pb);      // keys [16,32)
  }
}

__global__ void diff_attn(const unsigned short* __restrict__ qb,
                          const unsigned short* __restrict__ kb,
                          const unsigned short* __restrict__ vt,
                          const float* __restrict__ lq1, const float* __restrict__ lk1,
                          const float* __restrict__ lq2, const float* __restrict__ lk2,
                          const float* __restrict__ gamma, const float* __restrict__ beta,
                          float* __restrict__ out) {
  __shared__ alignas(16) unsigned short Kst[2][32 * 32];  // double-buffered K tile
  __shared__ alignas(16) unsigned short P1s[16 * 32];
  __shared__ alignas(16) unsigned short P2s[16 * 32];

  const int lane = threadIdx.x;
  const int lm = lane & 15, half = lane >> 4;
  const int nt = blockIdx.x;     // query tile 0..31
  const int h  = blockIdx.y;     // head
  const int bt = blockIdx.z;     // fused (b,t)

  // lambda = exp(lq1.lk1) - exp(lq2.lk2) + lambda_init  (computed in-wave)
  float p1 = (lane < 16) ? lq1[lane] * lk1[lane] : 0.f;
  float p2 = (lane < 16) ? lq2[lane] * lk2[lane] : 0.f;
  float lam = __expf(radd32(p1)) - __expf(radd32(p2)) + kLambdaInit;

  const float g_a = gamma[lm], g_b = gamma[16 + lm];
  const float b_a = beta[lm],  b_b = beta[16 + lm];

  // Q tiles as WMMA A operands, K padded to 32 with zeros in the upper half.
  const int qrow = bt * 512 + nt * 16 + lm;
  const unsigned short* qbase = qb + (size_t)qrow * 256 + h * 32;
  v8bf z8 = zero8();
  v16bf A1 = (half == 0) ? ld16(qbase) : cat8(z8, z8);        // q1: d in [0,16)
  v16bf A2 = (half == 0) ? ld16(qbase + 16) : cat8(z8, z8);   // q2: d in [16,32)
  v16bf z16 = cat8(z8, z8);

  v8f zacc = {0.f, 0.f, 0.f, 0.f, 0.f, 0.f, 0.f, 0.f};
  v8f a11 = zacc, a12 = zacc, a21 = zacc, a22 = zacc;
  float m1[8], l1[8], m2[8], l2[8];
#pragma unroll
  for (int r = 0; r < 8; ++r) { m1[r] = -1e30f; l1[r] = 0.f; m2[r] = -1e30f; l2[r] = 0.f; }

  const unsigned short* khead = kb + (size_t)(bt * 512) * 256 + h * 32;
  const unsigned short* vhead = vt + (((size_t)(bt * 8 + h)) * 32) * 512;
  const unsigned int kst_lds = (unsigned int)(uintptr_t)&Kst[0][0];

  // stage K tile 0 (keys [0,32), 32 channels) via TDM
#if HAVE_TDM
  tdm_load_tile_2d(kst_lds, khead);
#else
  { // fallback: one 64B K-row per lane
    const unsigned short* src = khead + (size_t)lane * 256;
    unsigned short* dstr = &Kst[0][lane * 32];
    ((uint4*)dstr)[0] = ((const uint4*)src)[0];
    ((uint4*)dstr)[1] = ((const uint4*)src)[1];
    ((uint4*)dstr)[2] = ((const uint4*)src)[2];
    ((uint4*)dstr)[3] = ((const uint4*)src)[3];
  }
#endif

  for (int mt = 0; mt < 16; ++mt) {
    // kick off next K tile, then wait for the current one (in-order TENSORcnt)
    if (mt + 1 < 16) {
#if HAVE_TDM
      tdm_load_tile_2d(kst_lds + ((mt + 1) & 1) * 2048,
                       khead + (size_t)(mt + 1) * 32 * 256);
      tdm_wait<1>();
#else
      const unsigned short* src = khead + (size_t)((mt + 1) * 32 + lane) * 256;
      unsigned short* dstr = &Kst[(mt + 1) & 1][lane * 32];
      ((uint4*)dstr)[0] = ((const uint4*)src)[0];
      ((uint4*)dstr)[1] = ((const uint4*)src)[1];
      ((uint4*)dstr)[2] = ((const uint4*)src)[2];
      ((uint4*)dstr)[3] = ((const uint4*)src)[3];
#endif
    } else {
      tdm_wait<0>();
    }

    // K rows as WMMA B from LDS (lanes 0..15 real data, upper K-half zero pad)
    const unsigned short* ktile = &Kst[mt & 1][0];
    v16bf Bk1a = z16, Bk1b = z16, Bk2a = z16, Bk2b = z16;
    if (half == 0) {
      Bk1a = ld16(ktile + lm * 32);             // keys [0,16), ch [0,16)
      Bk1b = ld16(ktile + (16 + lm) * 32);      // keys [16,32), ch [0,16)
      Bk2a = ld16(ktile + lm * 32 + 16);        // keys [0,16), ch [16,32)
      Bk2b = ld16(ktile + (16 + lm) * 32 + 16); // keys [16,32), ch [16,32)
    }

    v8f s1a = wmma_bf16(A1, Bk1a, zacc);
    v8f s1b = wmma_bf16(A1, Bk1b, zacc);
    v8f s2a = wmma_bf16(A2, Bk2a, zacc);
    v8f s2b = wmma_bf16(A2, Bk2b, zacc);

    flash_update(s1a, s1b, m1, l1, a11, a12, P1s, lm, half);
    flash_update(s2a, s2b, m2, l2, a21, a22, P2s, lm, half);

    // P tiles back as WMMA A (16 x 32 over this key block) via LDS transpose
    const unsigned short* p1row = P1s + lm * 32;
    const unsigned short* p2row = P2s + lm * 32;
    v16bf Ap1 = ld16(p1row + 16 * half);
    v16bf Ap2 = ld16(p2row + 16 * half);

    // V as WMMA B: lane owns channel d, 16 contiguous keys per K-half
    const unsigned short* vb = vhead + mt * 32 + 16 * half;
    v16bf Bv1 = ld16(vb + (size_t)lm * 512);          // d in [0,16)
    v16bf Bv2 = ld16(vb + (size_t)(16 + lm) * 512);   // d in [16,32)
    if (mt + 1 < 16)
      __builtin_prefetch(vhead + (mt + 1) * 32 + (size_t)lm * 512, 0, 1);

    a11 = wmma_bf16(Ap1, Bv1, a11);
    a12 = wmma_bf16(Ap1, Bv2, a12);
    a21 = wmma_bf16(Ap2, Bv1, a21);
    a22 = wmma_bf16(Ap2, Bv2, a22);
  }

  // epilogue: y = attn1 - lam*attn2, LayerNorm over 32 channels, scale, store
  const float oscale = 1.f - kLambdaInit;
#pragma unroll
  for (int r = 0; r < 8; ++r) {
    float i1 = 1.f / l1[r];
    float i2 = 1.f / l2[r];
    float ya = a11[r] * i1 - lam * (a21[r] * i2);   // channel lm
    float yb = a12[r] * i1 - lam * (a22[r] * i2);   // channel 16+lm
    float mu = radd16(ya + yb) * (1.f / 32.f);
    float da = ya - mu, db = yb - mu;
    float var = radd16(da * da + db * db) * (1.f / 32.f);
    float rstd = rsqrtf(var + kLnEps);
    float oa = (da * rstd * g_a + b_a) * oscale;
    float ob = (db * rstd * g_b + b_b) * oscale;
    int n = nt * 16 + r + 8 * half;
    float* orow = out + ((size_t)(bt * 512 + n)) * 256 + h * 32;
    orow[lm] = oa;
    orow[16 + lm] = ob;
  }
}

// ---- launcher --------------------------------------------------------------

extern "C" void kernel_launch(void* const* d_in, const int* in_sizes, int n_in,
                              void* d_out, int out_size, void* d_ws, size_t ws_size,
                              hipStream_t stream) {
  const float* x   = (const float*)d_in[0];
  const float* Wq  = (const float*)d_in[1];
  const float* Wk  = (const float*)d_in[2];
  const float* Wv  = (const float*)d_in[3];
  const float* lq1 = (const float*)d_in[4];
  const float* lk1 = (const float*)d_in[5];
  const float* lq2 = (const float*)d_in[6];
  const float* lk2 = (const float*)d_in[7];
  const float* gam = (const float*)d_in[8];
  const float* bet = (const float*)d_in[9];
  float* out = (float*)d_out;

  char* ws = (char*)d_ws;
  unsigned short* xb = (unsigned short*)ws;  ws += (size_t)MROWS_ * 256 * 2; // 8 MB
  unsigned short* wt = (unsigned short*)ws;  ws += (size_t)3 * 65536 * 2;    // 384 KB
  unsigned short* qb = (unsigned short*)ws;  ws += (size_t)MROWS_ * 256 * 2; // 8 MB
  unsigned short* kb = (unsigned short*)ws;  ws += (size_t)MROWS_ * 256 * 2; // 8 MB
  unsigned short* vt = (unsigned short*)ws;                                  // 8 MB

  // stage 0: bf16 conversion / weight transpose
  cvt_x_bf16<<<(MROWS_ * 256 / 4 + 255) / 256, 256, 0, stream>>>(x, xb, MROWS_ * 256 / 4);
  xpose_w_bf16<<<(3 * 65536 + 255) / 256, 256, 0, stream>>>(Wq, Wk, Wv, wt);

  // stage 1: QKV projection (bf16 WMMA)
  qkv_gemm<<<dim3(MROWS_ / 16, 16, 3), 32, 0, stream>>>(xb, wt, qb, kb, vt);

  // stage 2: differential attention + LayerNorm (TDM-staged K tiles)
  diff_attn<<<dim3(NTOK_ / 16, HEADS_, BT_), 32, 0, stream>>>(
      qb, kb, vt, lq1, lk1, lq2, lk2, gam, bet, out);
}
